// SynthesizerAttention_47940424958285
// MI455X (gfx1250) — compile-verified, matrix operand order checked by reading, not run on hardware
//
#include <hip/hip_runtime.h>

// ---------------------------------------------------------------------------
// Synthesizer attention for MI455X (gfx1250, wave32, WMMA f32_16x16x32_f16).
// Pipeline:
//   K1: r  = relu(x @ W1^T + b1)            (f16, row-major [4096,1024])
//   K2: vT = (x @ Wv^T + bv) transposed     (f16, [b,h,d,t] = [2,16,64,2048])
//   K3: w2t = w2^T                          (f16, [2048,64])
//   K4: fused causal softmax attention      (flash-style, never materializes att)
//   K5: out = y @ Wp^T + bp                 (fp32; f16 A tile staged to LDS via
//       global_load_async_to_lds_b128 + s_wait_asynccnt)
// ---------------------------------------------------------------------------

typedef __attribute__((ext_vector_type(16))) _Float16 v16h;
typedef __attribute__((ext_vector_type(8)))  _Float16 v8h;
typedef __attribute__((ext_vector_type(4)))  _Float16 v4h;
typedef __attribute__((ext_vector_type(8)))  float    v8f;

#define NB 2
#define TT 2048
#define CC 1024
#define NH 16
#define HS 64
#define MM (NB*TT)   // 4096

// Load a 16x32 f16 WMMA fragment (A-layout; also valid for N-major B) from a
// row-major matrix with leading dimension `ld` (halfs). Per CDNA5 ISA layout:
// lanes 0-15 hold rows 0-15 with K=0..7 (VGPR0-3) and K=16..23 (VGPR4-7);
// lanes 16-31 hold the same rows with K=8..15 and K=24..31.
__device__ __forceinline__ v16h load_frag16(const _Float16* p, int ld) {
  const int lane = threadIdx.x & 31;
  const int r  = lane & 15;
  const int kg = (lane >> 4) << 3;            // 0 or 8
  const _Float16* b = p + (size_t)r * ld + kg;
  v8h lo = *(const v8h*)(b);                  // K = kg .. kg+7
  v8h hi = *(const v8h*)(b + 16);             // K = kg+16 .. kg+23
  v16h out;
#pragma unroll
  for (int i = 0; i < 8; ++i) { out[i] = lo[i]; out[i + 8] = hi[i]; }
  return out;
}

__device__ __forceinline__ float hmax16(float v) {
#pragma unroll
  for (int m = 8; m >= 1; m >>= 1) v = fmaxf(v, __shfl_xor(v, m, 32));
  return v;
}
__device__ __forceinline__ float hsum16(float v) {
#pragma unroll
  for (int m = 8; m >= 1; m >>= 1) v += __shfl_xor(v, m, 32);
  return v;
}

// ---------------------------------------------------------------------------
// Generic GEMM: out = act(A @ W^T + bias).  A: [M,K] (f32 or f16), W: [N,K] f32.
// Block tile 128x64, BK=32, 8 waves; wave w computes rows [w*16, w*16+16).
// f16-A tiles are staged with async global->LDS DMA (no VGPR round-trip);
// fp32 operands need a cvt pass so they stage through VGPRs.
// ---------------------------------------------------------------------------
template<bool RELU, bool OUT_F32, bool STORE_TRANS, bool A_HALF>
__global__ __launch_bounds__(256)
void gemm_awt(const void* __restrict__ Av, const float* __restrict__ W,
              const float* __restrict__ bias, void* __restrict__ outv,
              int M, int N, int K) {
  __shared__ _Float16 As[128 * 40];   // 32 + 8 halfs pad (keeps 16B alignment)
  __shared__ _Float16 Bs[64 * 40];
  const int tid  = threadIdx.x;
  const int lane = tid & 31;
  const int wv   = tid >> 5;                       // 0..7
  const int row0 = blockIdx.y * 128;
  const int col0 = blockIdx.x * 64;

  v8f acc[4] = {};

  for (int kk = 0; kk < K; kk += 32) {
    // ---- stage A tile (128x32) into LDS as f16 ----
    if constexpr (A_HALF) {
      const _Float16* A = (const _Float16*)Av;
#pragma unroll
      for (int j = 0; j < 2; ++j) {
        int idx = tid + j * 256;                   // 0..511
        int r = idx >> 2, c = (idx & 3) << 3;
        // Async DMA 16B global -> LDS (ASYNCcnt). Low 32 bits of a generic
        // LDS pointer are the wave-relative LDS byte address (ISA 10.2).
        unsigned long long gaddr =
            (unsigned long long)(uintptr_t)(A + (size_t)(row0 + r) * K + kk + c);
        unsigned ldsoff = (unsigned)(uintptr_t)(&As[r * 40 + c]);
        asm volatile("global_load_async_to_lds_b128 %0, %1, off"
                     :: "v"(ldsoff), "v"(gaddr) : "memory");
      }
    } else {
      const float* A = (const float*)Av;
#pragma unroll
      for (int j = 0; j < 4; ++j) {
        int idx = tid + j * 256;                   // 0..1023
        int r = idx >> 3, c = (idx & 7) << 2;
        float4 t = *(const float4*)(A + (size_t)(row0 + r) * K + kk + c);
        v4h h; h[0] = (_Float16)t.x; h[1] = (_Float16)t.y;
               h[2] = (_Float16)t.z; h[3] = (_Float16)t.w;
        *(v4h*)(&As[r * 40 + c]) = h;
      }
    }
    // ---- stage B tile: rows of W (output columns), 64x32 ----
#pragma unroll
    for (int j = 0; j < 2; ++j) {
      int idx = tid + j * 256;                     // 0..511
      int r = idx >> 3, c = (idx & 7) << 2;
      float4 t = *(const float4*)(W + (size_t)(col0 + r) * K + kk + c);
      v4h h; h[0] = (_Float16)t.x; h[1] = (_Float16)t.y;
             h[2] = (_Float16)t.z; h[3] = (_Float16)t.w;
      *(v4h*)(&Bs[r * 40 + c]) = h;
    }
    if (kk + 32 < K) {   // hint next K tiles into cache (global_prefetch_b8)
      const size_t esz = A_HALF ? 2 : 4;
      __builtin_prefetch((const char*)Av +
          ((size_t)(row0 + (tid & 127)) * K + kk + 32) * esz, 0, 1);
      __builtin_prefetch((const char*)W +
          ((size_t)(col0 + (tid & 63)) * K + kk + 32) * 4, 0, 1);
    }
    if constexpr (A_HALF) {
      asm volatile("s_wait_asynccnt 0" ::: "memory");  // our DMA into LDS done
    }
    __syncthreads();

    v16h afrag = load_frag16(&As[(wv * 16) * 40], 40);
#pragma unroll
    for (int f = 0; f < 4; ++f) {
      v16h bfrag = load_frag16(&Bs[(f * 16) * 40], 40);
      acc[f] = __builtin_amdgcn_wmma_f32_16x16x32_f16(
          false, afrag, false, bfrag, (short)0, acc[f], false, false);
    }
    __syncthreads();
  }

  // ---- epilogue: bias, activation, store ----
  const int n  = lane & 15;
  const int m0 = (lane >> 4) << 3;
#pragma unroll
  for (int f = 0; f < 4; ++f) {
    const int gc = col0 + f * 16 + n;
    const float bval = bias[gc];
#pragma unroll
    for (int g = 0; g < 8; ++g) {
      const int gr = row0 + wv * 16 + m0 + g;
      float v = acc[f][g] + bval;
      if constexpr (RELU) v = fmaxf(v, 0.0f);
      if constexpr (OUT_F32) {
        ((float*)outv)[(size_t)gr * N + gc] = v;
      } else if constexpr (STORE_TRANS) {
        // store v-projection transposed per head: vT[b,h,d,t]
        const int hh = gc >> 6, d = gc & 63;
        const int bb = gr >> 11, t = gr & 2047;     // T = 2048
        ((_Float16*)outv)[(((size_t)(bb * NH + hh) * HS + d) << 11) + t] =
            (_Float16)v;
      } else {
        ((_Float16*)outv)[(size_t)gr * N + gc] = (_Float16)v;
      }
    }
  }
}

// w2 [64,2048] f32 -> w2t [2048,64] f16
__global__ __launch_bounds__(256)
void w2_transpose(const float* __restrict__ w2, _Float16* __restrict__ w2t) {
  int idx  = blockIdx.x * 256 + threadIdx.x;      // 0..131071
  int nrow = idx >> 6, k = idx & 63;
  w2t[idx] = (_Float16)w2[(size_t)k * TT + nrow];
}

// ---------------------------------------------------------------------------
// Fused causal synthesizer attention. One wave owns a 16-row query tile.
// Online softmax over 32-column key tiles; P converted C-layout -> A-layout
// through a private LDS tile (same-wave RAW guarded by s_wait_dscnt).
// ---------------------------------------------------------------------------
__global__ __launch_bounds__(128)
void synth_attention(const _Float16* __restrict__ r, const _Float16* __restrict__ w2t,
                     const float* __restrict__ b2, const _Float16* __restrict__ vT,
                     _Float16* __restrict__ y) {
  __shared__ _Float16 P[4][16 * 40];
  const int lane = threadIdx.x & 31;
  const int wv   = threadIdx.x >> 5;
  const int gw   = blockIdx.x * 4 + wv;           // 0..4095 wave tiles
  const int tile = gw & 127;                      // T/16 tiles per head
  const int hg   = gw >> 7;
  const int b    = hg >> 4;
  const int h    = hg & 15;
  const int t0   = tile << 4;

  const int n  = lane & 15;
  const int m0 = (lane >> 4) << 3;

  // A fragments: 16 query rows x 64 head dims (two K=32 fragments)
  const _Float16* rbase = r + (size_t)(b * TT + t0) * CC + h * HS;
  const v16h ra0 = load_frag16(rbase, CC);
  const v16h ra1 = load_frag16(rbase + 32, CC);

  const _Float16* vtb = vT + (size_t)(b * NH + h) * HS * TT;

  v8f yacc[4] = {};
  float mrun[8], srun[8];
#pragma unroll
  for (int g = 0; g < 8; ++g) { mrun[g] = -3.0e38f; srun[g] = 0.0f; }

  for (int j0 = 0; j0 < t0 + 16; j0 += 32) {      // causal: wave-uniform bound
    // ---- scores S[16 x 32] = r_tile @ w2[:, j0:j0+32] ----
    v8f s0 = {}, s1 = {};
    {
      v16h b00 = load_frag16(w2t + (size_t)j0 * HS, HS);
      v16h b01 = load_frag16(w2t + (size_t)j0 * HS + 32, HS);
      s0 = __builtin_amdgcn_wmma_f32_16x16x32_f16(false, ra0, false, b00, (short)0, s0, false, false);
      s0 = __builtin_amdgcn_wmma_f32_16x16x32_f16(false, ra1, false, b01, (short)0, s0, false, false);
      v16h b10 = load_frag16(w2t + (size_t)(j0 + 16) * HS, HS);
      v16h b11 = load_frag16(w2t + (size_t)(j0 + 16) * HS + 32, HS);
      s1 = __builtin_amdgcn_wmma_f32_16x16x32_f16(false, ra0, false, b10, (short)0, s1, false, false);
      s1 = __builtin_amdgcn_wmma_f32_16x16x32_f16(false, ra1, false, b11, (short)0, s1, false, false);
    }
    const float bia0 = b2[j0 + n];
    const float bia1 = b2[j0 + 16 + n];

    _Float16* Pw = P[wv];
#pragma unroll
    for (int g = 0; g < 8; ++g) {
      const int trow = t0 + m0 + g;
      float v0 = s0[g] + bia0; if (j0 + n      > trow) v0 = -1.0e30f;
      float v1 = s1[g] + bia1; if (j0 + 16 + n > trow) v1 = -1.0e30f;
      const float tm = hmax16(fmaxf(v0, v1));
      const float nm = fmaxf(mrun[g], tm);
      const float scale = __expf(mrun[g] - nm);
      const float p0 = __expf(v0 - nm);
      const float p1 = __expf(v1 - nm);
      srun[g] = srun[g] * scale + hsum16(p0 + p1);
      mrun[g] = nm;
#pragma unroll
      for (int f = 0; f < 4; ++f) yacc[f][g] *= scale;
      Pw[(m0 + g) * 40 + n]      = (_Float16)p0;
      Pw[(m0 + g) * 40 + n + 16] = (_Float16)p1;
    }
    // same-wave LDS RAW: DS pipe is in-order per wave, but make it explicit
    asm volatile("s_wait_dscnt 0" ::: "memory");

    const v16h pf = load_frag16(Pw, 40);          // P as A-fragment [16 x 32]
#pragma unroll
    for (int f = 0; f < 4; ++f) {
      v16h vf = load_frag16(vtb + (size_t)(f * 16) * TT + j0, TT);
      yacc[f] = __builtin_amdgcn_wmma_f32_16x16x32_f16(
          false, pf, false, vf, (short)0, yacc[f], false, false);
    }
  }

  // ---- normalize and store y (f16) ----
  _Float16* yrow = y + (size_t)(b * TT + t0) * CC + h * HS;
#pragma unroll
  for (int f = 0; f < 4; ++f) {
#pragma unroll
    for (int g = 0; g < 8; ++g) {
      yrow[(size_t)(m0 + g) * CC + f * 16 + n] =
          (_Float16)(yacc[f][g] / srun[g]);
    }
  }
}

// ---------------------------------------------------------------------------
extern "C" void kernel_launch(void* const* d_in, const int* in_sizes, int n_in,
                              void* d_out, int out_size, void* d_ws, size_t ws_size,
                              hipStream_t stream) {
  (void)in_sizes; (void)n_in; (void)out_size; (void)ws_size;
  const float* x  = (const float*)d_in[0];
  const float* W1 = (const float*)d_in[1];
  const float* b1 = (const float*)d_in[2];
  const float* w2 = (const float*)d_in[3];
  const float* b2 = (const float*)d_in[4];
  const float* Wv = (const float*)d_in[5];
  const float* bv = (const float*)d_in[6];
  const float* Wp = (const float*)d_in[7];
  const float* bp = (const float*)d_in[8];
  float* out = (float*)d_out;

  // Workspace layout (needs ~24.5 MB): r, vT, y (f16, 8 MB each) + w2t (256 KB)
  _Float16* r   = (_Float16*)d_ws;
  _Float16* vT  = r  + (size_t)MM * CC;
  _Float16* yh  = vT + (size_t)MM * CC;
  _Float16* w2t = yh + (size_t)MM * CC;

  dim3 gg(CC / 64, MM / 128);   // 16 x 32 blocks

  // K1: r = relu(x @ W1^T + b1)
  gemm_awt<true,  false, false, false><<<gg, 256, 0, stream>>>(x, W1, b1, r, MM, CC, CC);
  // K2: vT = transpose_per_head(x @ Wv^T + bv)
  gemm_awt<false, false, true,  false><<<gg, 256, 0, stream>>>(x, Wv, bv, vT, MM, CC, CC);
  // K3: w2t = w2^T (f16)
  w2_transpose<<<(TT * HS) / 256, 256, 0, stream>>>(w2, w2t);
  // K4: fused causal attention -> y (f16)
  synth_attention<<<(NB * NH * (TT / 16)) / 4, 128, 0, stream>>>(r, w2t, b2, vT, yh);
  // K5: out = y @ Wp^T + bp (fp32)
  gemm_awt<false, true,  false, true ><<<gg, 256, 0, stream>>>(yh, Wp, bp, out, MM, CC, CC);
}
